// ProvenanceGNNv4_28879360098528
// MI455X (gfx1250) — compile-verified, matrix operand order checked by reading, not run on hardware
//
#include <hip/hip_runtime.h>
#include <hip/hip_bf16.h>

// ---------------------------------------------------------------------------
// Types for CDNA5 WMMA (wave32, 16x16x32 f16 -> f32 accumulate)
// ---------------------------------------------------------------------------
typedef __attribute__((ext_vector_type(16))) _Float16 v16h;
typedef __attribute__((ext_vector_type(8)))  float    v8f;
typedef __attribute__((ext_vector_type(4)))  unsigned v4u;
typedef __attribute__((ext_vector_type(4)))  unsigned u32x4;
typedef __attribute__((ext_vector_type(8)))  int      i32x8;
typedef __attribute__((ext_vector_type(4)))  int      i32x4;

union AFrag { v16h v; unsigned u[8]; };
union BFrag { v16h v; v4u q[2]; unsigned u[8]; };
union Pack2 { unsigned u; _Float16 h[2]; };

#if defined(__gfx1250__) && __has_builtin(__builtin_amdgcn_tensor_load_to_lds) && \
    __has_builtin(__builtin_amdgcn_s_wait_tensorcnt)
#define USE_TDM 1
#else
#define USE_TDM 0
#endif

// K-offset base for the documented 16-bit A-matrix 16x32 layout (CDNA5 ISA):
//   lanes 0-15: V0..V3 -> K=0..7, V4..V7 -> K=16..23 ; lanes 16-31: +8
__device__ __forceinline__ int kbase32(int lane, int d) {
    return ((d & 3) << 1) + ((d >> 2) << 4) + ((lane >> 4) << 3);
}

// One 16-row x 16-col output tile, K = KP, A staged in LDS (row-major halves),
// B pre-packed in global memory in fragment order [kt][nt][lane][dword].
// NT = number of 16-col tiles the packed weight matrix holds.
template <int KP, int NT>
__device__ __forceinline__ v8f wmma_rowtile(const unsigned* __restrict__ sA,
                                            const unsigned* __restrict__ wfrag,
                                            int lane, int nt, v8f acc) {
    const int m = lane & 15;
    const int rowbase = m * (KP >> 1);            // dwords per row = KP/2
#pragma unroll
    for (int kt = 0; kt < KP / 32; ++kt) {
        AFrag a; BFrag b;
#pragma unroll
        for (int d = 0; d < 8; ++d) {
            int kb = kt * 32 + kbase32(lane, d);  // even
            a.u[d] = sA[rowbase + (kb >> 1)];
        }
        const v4u* bp = (const v4u*)(wfrag + (((size_t)(kt * NT + nt) * 32 + lane) << 3));
        b.q[0] = bp[0];
        b.q[1] = bp[1];
        acc = __builtin_amdgcn_wmma_f32_16x16x32_f16(false, a.v, false, b.v,
                                                     (short)0, acc, false, false);
    }
    return acc;
}

// ---------------------------------------------------------------------------
// Weight prep: f32 row-major W[K][ncol] -> f16 B-fragment layout (zero-padded)
// ---------------------------------------------------------------------------
__global__ void k_prep_wfrag(const float* __restrict__ W, unsigned* __restrict__ dst,
                             int Ksrc, int Kp, int ncol) {
    int idx = blockIdx.x * 256 + threadIdx.x;
    int ntcnt = ncol >> 4;
    int total = (Kp >> 1) * ncol;                 // dwords
    if (idx >= total) return;
    int d    = idx & 7;
    int lane = (idx >> 3) & 31;
    int t    = idx >> 8;
    int nt   = t % ntcnt;
    int kt   = t / ntcnt;
    int n  = nt * 16 + (lane & 15);
    int kb = kt * 32 + kbase32(lane, d);
    float lo = (kb     < Ksrc) ? W[(size_t)kb * ncol + n]       : 0.f;
    float hi = (kb + 1 < Ksrc) ? W[(size_t)(kb + 1) * ncol + n] : 0.f;
    Pack2 p; p.h[0] = (_Float16)lo; p.h[1] = (_Float16)hi;
    dst[idx] = p.u;
}

__global__ void k_zero(float* p, size_t n) {
    size_t i = (size_t)blockIdx.x * blockDim.x + threadIdx.x;
    size_t s = (size_t)gridDim.x * blockDim.x;
    for (; i < n; i += s) p[i] = 0.f;
}

// ---------------------------------------------------------------------------
// h = relu(bn(x @ W_in + b))        [N,64] -> [N,128]
// ---------------------------------------------------------------------------
__global__ __launch_bounds__(256) void k_input(
    const float* __restrict__ x, const unsigned* __restrict__ wfrag,
    const float* __restrict__ b, const float* __restrict__ bng,
    const float* __restrict__ bnb, const float* __restrict__ bnm,
    const float* __restrict__ bnv, float* __restrict__ h, int N_) {
    __shared__ unsigned sA[16 * 32];              // 16 rows x 64 halves
    int base = blockIdx.x * 16, tid = threadIdx.x;
    bool full = (base + 16 <= N_);
    _Float16* sAh = (_Float16*)sA;
    for (int i = tid; i < 16 * 64; i += 256) {
        int r = i >> 6, k = i & 63, node = base + r;
        sAh[i] = (_Float16)((full || node < N_) ? x[(size_t)node * 64 + k] : 0.f);
    }
    __syncthreads();
    int wave = tid >> 5, lane = tid & 31;
    v8f acc = {};
    acc = wmma_rowtile<64, 8>(sA, wfrag, lane, wave, acc);
    int n = wave * 16 + (lane & 15);
    float bias = b[n], g = bng[n], bb = bnb[n], m = bnm[n];
    float rv = rsqrtf(bnv[n] + 1e-5f);
    if (full) {
#pragma unroll
        for (int j = 0; j < 8; ++j) {
            int r = j + ((lane >> 4) << 3);
            float v = (acc[j] + bias - m) * rv * g + bb;
            h[(size_t)(base + r) * 128 + n] = fmaxf(v, 0.f);
        }
    } else {
#pragma unroll
        for (int j = 0; j < 8; ++j) {
            int r = j + ((lane >> 4) << 3), node = base + r;
            if (node < N_) {
                float v = (acc[j] + bias - m) * rv * g + bb;
                h[(size_t)node * 128 + n] = fmaxf(v, 0.f);
            }
        }
    }
}

// ---------------------------------------------------------------------------
// e = relu(edge_attr @ W_e + b)     [E,21] -> [E,128] f16 (K padded to 32)
// ---------------------------------------------------------------------------
__global__ __launch_bounds__(256) void k_edge_embed(
    const float* __restrict__ ea, const unsigned* __restrict__ wfrag,
    const float* __restrict__ b, _Float16* __restrict__ e16, int E_) {
    __shared__ unsigned sA[16 * 16];              // 16 x 32 halves
    int base = blockIdx.x * 16, tid = threadIdx.x;
    bool full = (base + 16 <= E_);
    _Float16* sAh = (_Float16*)sA;
    for (int i = tid; i < 16 * 32; i += 256) {
        int r = i >> 5, k = i & 31, e = base + r;
        sAh[i] = (_Float16)(((full || e < E_) && k < 21) ? ea[(size_t)e * 21 + k] : 0.f);
    }
    __syncthreads();
    int wave = tid >> 5, lane = tid & 31;
    v8f acc = {};
    acc = wmma_rowtile<32, 8>(sA, wfrag, lane, wave, acc);
    int n = wave * 16 + (lane & 15);
    float bias = b[n];
#pragma unroll
    for (int j = 0; j < 8; ++j) {
        int r = j + ((lane >> 4) << 3), e = base + r;
        if (full || e < E_)
            e16[(size_t)e * 128 + n] = (_Float16)fmaxf(acc[j] + bias, 0.f);
    }
}

// ---------------------------------------------------------------------------
// Fused per-layer edge pipeline:
//   ee = e @ W_le + b ; msg = relu(h[src] + ee) ; aggr[dst] += msg
// e-tile staged into LDS by the Tensor Data Mover when available; h[src]
// gather rows prefetched before the WMMA loop so DRAM latency hides under it.
// ---------------------------------------------------------------------------
__global__ __launch_bounds__(256) void k_edge_msg(
    const _Float16* __restrict__ e16, const unsigned* __restrict__ wfrag,
    const float* __restrict__ b, const int* __restrict__ src,
    const int* __restrict__ dst, const float* __restrict__ h,
    float* __restrict__ aggr, int E_) {
    __shared__ unsigned sA[16 * 64];              // 16 x 128 halves (4KB tile)
    int base = blockIdx.x * 16, tid = threadIdx.x;
    bool full = (base + 16 <= E_);
    int wave = tid >> 5, lane = tid & 31;
    int n = wave * 16 + (lane & 15);

#if USE_TDM
    if (tid < 32) {                               // one wave issues the DMA
        unsigned long long ga = (unsigned long long)(e16 + (size_t)base * 128);
        unsigned lds_off = (unsigned)(unsigned long long)(const void*)sA;
        unsigned rows = full ? 16u : (unsigned)(E_ - base);
        // D# group0: count=1, lds_addr, global_addr, type=2  (CDNA5 ISA)
        u32x4 g0;
        g0[0] = 1u;
        g0[1] = lds_off;
        g0[2] = (unsigned)(ga & 0xffffffffull);
        g0[3] = (unsigned)((ga >> 32) & 0x1ffffffull) | (2u << 30);
        // D# group1: data_size=8B, tensor 32 x rows (units), stride0=32,
        // tile 32 x rows -> one contiguous 256B row per edge  (§8.4)
        unsigned dim0 = 32u, dim1 = rows, tile0 = 32u, tile1 = rows, str0 = 32u;
        i32x8 g1;
        g1[0] = (int)(3u << 16);                              // data_size = 3 (8B)
        g1[1] = (int)((dim0 & 0xffffu) << 16);                // tensor_dim0 lo16
        g1[2] = (int)((dim0 >> 16) | ((dim1 & 0xffffu) << 16));
        g1[3] = (int)((dim1 >> 16) | (tile0 << 16));
        g1[4] = (int)tile1;                                   // tile_dim1, tile_dim2=0
        g1[5] = (int)str0;                                    // tensor_dim0_stride lo32
        g1[6] = 0;
        g1[7] = 0;
        i32x4 gz4 = {0, 0, 0, 0};                             // 2D: groups 2/3 unused
        i32x8 gz8 = {0, 0, 0, 0, 0, 0, 0, 0};
        __builtin_amdgcn_tensor_load_to_lds(g0, g1, gz4, gz4, gz8, 0);
        __builtin_amdgcn_s_wait_tensorcnt(0);
    }
#else
    {
        const unsigned* eu = (const unsigned*)e16;
        for (int i = tid; i < 1024; i += 256) {
            int r = i >> 6, k = i & 63, e = base + r;
            sA[i] = (full || e < E_) ? eu[(size_t)e * 64 + k] : 0u;
        }
    }
#endif

    // Load gather/scatter indices early and prefetch the h[src] rows so the
    // scattered reads are L2/L0-resident by the time the epilogue needs them.
    int srcv[8], dstv[8];
#pragma unroll
    for (int j = 0; j < 8; ++j) {
        int r = j + ((lane >> 4) << 3);
        int e = base + r;
        e = (e < E_) ? e : (E_ - 1);
        srcv[j] = src[e];
        dstv[j] = dst[e];
        __builtin_prefetch(&h[(size_t)srcv[j] * 128 + n], 0, 3);
    }
    __syncthreads();

    v8f acc = {};
    acc = wmma_rowtile<128, 8>(sA, wfrag, lane, wave, acc);
    float bias = b[n];
    if (full) {
#pragma unroll
        for (int j = 0; j < 8; ++j) {
            float v = fmaxf(h[(size_t)srcv[j] * 128 + n] + acc[j] + bias, 0.f);
            atomicAdd(&aggr[(size_t)dstv[j] * 128 + n], v);
        }
    } else {
#pragma unroll
        for (int j = 0; j < 8; ++j) {
            int r = j + ((lane >> 4) << 3), e = base + r;
            if (e < E_) {
                float v = fmaxf(h[(size_t)srcv[j] * 128 + n] + acc[j] + bias, 0.f);
                atomicAdd(&aggr[(size_t)dstv[j] * 128 + n], v);
            }
        }
    }
}

// ---------------------------------------------------------------------------
// Fused node update: z=(1+eps)h+aggr ; t=relu(bn(z@W1+b1)) ; z2=t@W2+b2 ;
// z2=LN(z2) ; h=relu(z2+h) ; outs16[l]=h (f16 for JK)
// ---------------------------------------------------------------------------
__global__ __launch_bounds__(256) void k_node_update(
    float* __restrict__ h, const float* __restrict__ aggr,
    const float* __restrict__ epsp, const unsigned* __restrict__ wf1,
    const float* __restrict__ b1, const float* __restrict__ bng,
    const float* __restrict__ bnb, const float* __restrict__ bnm,
    const float* __restrict__ bnv, const unsigned* __restrict__ wf2,
    const float* __restrict__ b2, const float* __restrict__ lng,
    const float* __restrict__ lnb, _Float16* __restrict__ out16, int N_) {
    __shared__ unsigned sA[16 * 64];              // 16 x 128 halves (A for both GEMMs)
    __shared__ float    sZ[16 * 128];
    __shared__ float    sMu[16], sRs[16];
    int base = blockIdx.x * 16, tid = threadIdx.x;
    bool full = (base + 16 <= N_);
    float eps1 = 1.f + epsp[0];
    _Float16* sAh = (_Float16*)sA;
    for (int i = tid; i < 2048; i += 256) {
        int r = i >> 7, c = i & 127, node = base + r;
        float z = (full || node < N_)
                      ? eps1 * h[(size_t)node * 128 + c] + aggr[(size_t)node * 128 + c]
                      : 0.f;
        sAh[i] = (_Float16)z;
    }
    __syncthreads();
    int wave = tid >> 5, lane = tid & 31;
    int n = wave * 16 + (lane & 15);
    v8f acc = {};
    acc = wmma_rowtile<128, 8>(sA, wf1, lane, wave, acc);
    __syncthreads();                              // all reads of sA done
    {
        float bias = b1[n], g = bng[n], bb = bnb[n], m = bnm[n];
        float rv = rsqrtf(bnv[n] + 1e-5f);
#pragma unroll
        for (int j = 0; j < 8; ++j) {
            int r = j + ((lane >> 4) << 3);
            float t = (acc[j] + bias - m) * rv * g + bb;
            sAh[r * 128 + n] = (_Float16)fmaxf(t, 0.f);
        }
    }
    __syncthreads();
    v8f acc2 = {};
    acc2 = wmma_rowtile<128, 8>(sA, wf2, lane, wave, acc2);
    {
        float bias = b2[n];
#pragma unroll
        for (int j = 0; j < 8; ++j) {
            int r = j + ((lane >> 4) << 3);
            sZ[r * 128 + n] = acc2[j] + bias;
        }
    }
    __syncthreads();
    if (tid < 16) {                               // per-row LayerNorm stats
        float mu = 0.f;
        for (int c = 0; c < 128; ++c) mu += sZ[tid * 128 + c];
        mu *= (1.f / 128.f);
        float var = 0.f;
        for (int c = 0; c < 128; ++c) {
            float d = sZ[tid * 128 + c] - mu;
            var += d * d;
        }
        var *= (1.f / 128.f);
        sMu[tid] = mu;
        sRs[tid] = rsqrtf(var + 1e-5f);
    }
    __syncthreads();
    for (int i = tid; i < 2048; i += 256) {
        int r = i >> 7, c = i & 127, node = base + r;
        if (full || node < N_) {
            float z = (sZ[i] - sMu[r]) * sRs[r] * lng[c] + lnb[c];
            float hn = fmaxf(z + h[(size_t)node * 128 + c], 0.f);
            h[(size_t)node * 128 + c] = hn;
            out16[(size_t)node * 128 + c] = (_Float16)hn;
        }
    }
}

// ---------------------------------------------------------------------------
// Fused JK attention (WMMA) + softmax + sum/max/count pooling.
// Block = 16 nodes. 12 (layer, col-tile) WMMA tiles over jk1 [128->64];
// relu(t)*w2 reduced in LDS -> attention logits -> softmax -> pooling reuses
// the already-staged f16 layer tiles in LDS. batch is sorted -> LDS
// pre-reduction per graph, few global atomics.
// ---------------------------------------------------------------------------
__global__ __launch_bounds__(256) void k_jk_pool(
    const _Float16* __restrict__ outs, const int* __restrict__ batch,
    const unsigned* __restrict__ fJK1, const float* __restrict__ b1v,
    const float* __restrict__ w2, const float* __restrict__ b2s,
    float* __restrict__ psum, float* __restrict__ pmax,
    float* __restrict__ pcnt, int N_) {
    __shared__ unsigned sA[3 * 16 * 64];          // 3 layer tiles, 16x128 halves (12KB)
    __shared__ float    sP[3 * 16 * 64];          // relu(t)*w2 contributions (12KB)
    __shared__ float    sAtt[48];
    __shared__ float    sWgt[48];
    __shared__ float    sSum[128];
    __shared__ unsigned sMax[128];
    __shared__ int      sB[16];
    __shared__ int      sCnt, sG0, sG1;
    int tid = threadIdx.x, base = blockIdx.x * 16;
    size_t plane = (size_t)N_ * 64;               // dwords per layer plane
    const unsigned* ou = (const unsigned*)outs;
    _Float16* sAh = (_Float16*)sA;
    for (int i = tid; i < 3072; i += 256) {
        int l = i >> 10, r = (i >> 6) & 15, k = i & 63, node = base + r;
        sA[i] = (node < N_) ? ou[(size_t)l * plane + (size_t)node * 64 + k] : 0u;
    }
    if (tid < 16) sB[tid] = (base + tid < N_) ? batch[base + tid] : -1;
    if (tid == 0) {
        int lo = (base < N_) ? base : N_ - 1;
        int hi = (base + 15 < N_) ? base + 15 : N_ - 1;
        sG0 = batch[lo];
        sG1 = batch[hi];
    }
    __syncthreads();
    int wave = tid >> 5, lane = tid & 31;
    for (int T = wave; T < 12; T += 8) {          // (layer l, col-tile nt)
        int l = T >> 2, nt = T & 3;
        v8f acc = {};
        acc = wmma_rowtile<128, 4>(sA + l * 1024, fJK1, lane, nt, acc);
        int col = nt * 16 + (lane & 15);
        float bias = b1v[col], wj = w2[col];
#pragma unroll
        for (int j = 0; j < 8; ++j) {
            int r = j + ((lane >> 4) << 3);
            sP[(l * 16 + r) * 64 + col] = fmaxf(acc[j] + bias, 0.f) * wj;
        }
    }
    __syncthreads();
    if (tid < 48) {                               // a[l][row] = b2 + sum_j
        float s = b2s[0];
        for (int c = 0; c < 64; ++c) s += sP[tid * 64 + c];
        sAtt[tid] = s;
    }
    __syncthreads();
    if (tid < 16) {                               // softmax over 3 layers
        float a0 = sAtt[tid], a1 = sAtt[16 + tid], a2 = sAtt[32 + tid];
        float am = fmaxf(a0, fmaxf(a1, a2));
        float e0 = __expf(a0 - am), e1 = __expf(a1 - am), e2 = __expf(a2 - am);
        float inv = 1.f / (e0 + e1 + e2);
        sWgt[tid] = e0 * inv;
        sWgt[16 + tid] = e1 * inv;
        sWgt[32 + tid] = e2 * inv;
    }
    __syncthreads();
    int c = tid & 127, rh = tid >> 7;             // 2 rows x 128 cols per pass
    for (int gg = sG0; gg <= sG1; ++gg) {
        if (tid < 128) { sSum[tid] = 0.f; sMax[tid] = 0u; }
        if (tid == 0) sCnt = 0;
        __syncthreads();
        for (int rr = 0; rr < 8; ++rr) {
            int row = rr * 2 + rh;
            if (base + row < N_ && sB[row] == gg) {
                float hx = sWgt[row]      * (float)sAh[row * 128 + c]
                         + sWgt[16 + row] * (float)sAh[2048 + row * 128 + c]
                         + sWgt[32 + row] * (float)sAh[4096 + row * 128 + c];
                atomicAdd(&sSum[c], hx);
                atomicMax(&sMax[c], __float_as_uint(hx));   // hx >= 0
                if (c == 0) atomicAdd(&sCnt, 1);
            }
        }
        __syncthreads();
        if (tid < 128) {
            if (sSum[tid] != 0.f) atomicAdd(&psum[gg * 128 + tid], sSum[tid]);
            if (sMax[tid] != 0u) atomicMax((unsigned*)&pmax[gg * 128 + tid], sMax[tid]);
        }
        if (tid == 0 && sCnt > 0) atomicAdd(&pcnt[gg], (float)sCnt);
        __syncthreads();
    }
}

// ---------------------------------------------------------------------------
// Final classifier on G=64 graphs: emb=[mean|max|sum|gf] -> clf1/bn/relu/clf2
// ---------------------------------------------------------------------------
__global__ __launch_bounds__(128) void k_final(
    const float* __restrict__ psum, const float* __restrict__ pmax,
    const float* __restrict__ pcnt, const float* __restrict__ gfeat,
    const float* __restrict__ gfW, const float* __restrict__ gfB,
    const float* __restrict__ c1W, const float* __restrict__ c1B,
    const float* __restrict__ bg, const float* __restrict__ bb,
    const float* __restrict__ bm, const float* __restrict__ bv,
    const float* __restrict__ c2W, const float* __restrict__ c2B,
    float* __restrict__ out) {
    __shared__ float sEmb[416];
    __shared__ float sRed[128];
    int g = blockIdx.x, tid = threadIdx.x;
    float denom = fmaxf(pcnt[g], 1.f);
    for (int i = tid; i < 416; i += 128) {
        float v;
        if (i < 128)       v = psum[g * 128 + i] / denom;
        else if (i < 256)  v = pmax[g * 128 + (i - 128)];
        else if (i < 384)  v = psum[g * 128 + (i - 256)];
        else {
            int j = i - 384;
            v = gfB[j];
            for (int k = 0; k < 10; ++k) v += gfeat[g * 10 + k] * gfW[k * 32 + j];
        }
        sEmb[i] = v;
    }
    __syncthreads();
    int j = tid;
    float t = c1B[j];
    for (int k = 0; k < 416; ++k) t += sEmb[k] * c1W[k * 128 + j];
    t = (t - bm[j]) * rsqrtf(bv[j] + 1e-5f) * bg[j] + bb[j];
    t = fmaxf(t, 0.f);
    sRed[j] = t * c2W[j];
    __syncthreads();
    for (int off = 64; off > 0; off >>= 1) {
        if (tid < off) sRed[tid] += sRed[tid + off];
        __syncthreads();
    }
    if (tid == 0) out[g] = sRed[0] + c2B[0];
}

// ---------------------------------------------------------------------------
// Host side
// ---------------------------------------------------------------------------
// Input index map (setup_inputs() dict order, params flattened in insertion order):
// 0:x 1:edge_index 2:edge_attr 3:batch 4:graph_features
// 5..10: inp.{W,b,bn.g,bn.b,bn.m,bn.v}   11,12: edge.{W,b}
// blocks[l] @ 13+13*l: {eps, lin_e.W, lin_e.b, mlp1.W, mlp1.b,
//                       bn.g, bn.b, bn.m, bn.v, mlp2.W, mlp2.b, ln_g, ln_b}
// 52,53:jk1.{W,b} 54,55:jk2.{W,b} 56,57:gf.{W,b}
// 58,59:clf1.{W,b} 60..63:clf_bn.{g,b,m,v} 64,65:clf2.{W,b}
extern "C" void kernel_launch(void* const* d_in, const int* in_sizes, int n_in,
                              void* d_out, int out_size, void* d_ws, size_t ws_size,
                              hipStream_t stream) {
    (void)n_in; (void)out_size; (void)ws_size;
    const int N_ = in_sizes[3];          // 200000
    const int E_ = in_sizes[1] / 2;      // 600000
    const int G_ = in_sizes[4] / 10;     // 64

    const float* x     = (const float*)d_in[0];
    const int*   ei    = (const int*)d_in[1];
    const float* ea    = (const float*)d_in[2];
    const int*   batch = (const int*)d_in[3];
    const float* gfeat = (const float*)d_in[4];
    const int* src = ei, * dst = ei + E_;

    // workspace carve-out (256B aligned)
    char* ws = (char*)d_ws;
    size_t off = 0;
    auto carve = [&](size_t bytes) -> void* {
        void* p = ws + off;
        off = (off + bytes + 255) & ~(size_t)255;
        return p;
    };
    float*     h     = (float*)carve((size_t)N_ * 128 * 4);
    _Float16*  outs  = (_Float16*)carve((size_t)3 * N_ * 128 * 2);
    _Float16*  e16   = (_Float16*)carve((size_t)E_ * 128 * 2);
    float*     aggr  = (float*)carve((size_t)N_ * 128 * 4);
    float*     psum  = (float*)carve((size_t)G_ * 128 * 4);
    float*     pmax  = (float*)carve((size_t)G_ * 128 * 4);
    float*     pcnt  = (float*)carve((size_t)G_ * 4);
    unsigned*  fInp  = (unsigned*)carve(64 * 256);     // Kp=64,  ncol=128
    unsigned*  fEdge = (unsigned*)carve(32 * 256);     // Kp=32 (K=21 padded)
    unsigned*  fJK1  = (unsigned*)carve(128 * 128);    // Kp=128, ncol=64
    unsigned*  fLE[3], *fM1[3], *fM2[3];
    for (int l = 0; l < 3; ++l) {
        fLE[l] = (unsigned*)carve(128 * 256);
        fM1[l] = (unsigned*)carve(128 * 256);
        fM2[l] = (unsigned*)carve(128 * 256);
    }

    auto prep = [&](const void* W, unsigned* dstp, int Ksrc, int Kp, int ncol) {
        int total = (Kp >> 1) * ncol;
        k_prep_wfrag<<<(total + 255) / 256, 256, 0, stream>>>((const float*)W, dstp,
                                                              Ksrc, Kp, ncol);
    };
    prep(d_in[5], fInp, 64, 64, 128);
    prep(d_in[11], fEdge, 21, 32, 128);
    prep(d_in[52], fJK1, 128, 128, 64);
    for (int l = 0; l < 3; ++l) {
        int b0 = 13 + 13 * l;
        prep(d_in[b0 + 1], fLE[l], 128, 128, 128);
        prep(d_in[b0 + 3], fM1[l], 128, 128, 128);
        prep(d_in[b0 + 9], fM2[l], 128, 128, 128);
    }

    const int Ntiles = (N_ + 15) / 16;
    const int Etiles = (E_ + 15) / 16;

    k_input<<<Ntiles, 256, 0, stream>>>(x, fInp, (const float*)d_in[6],
        (const float*)d_in[7], (const float*)d_in[8], (const float*)d_in[9],
        (const float*)d_in[10], h, N_);
    k_edge_embed<<<Etiles, 256, 0, stream>>>(ea, fEdge, (const float*)d_in[12], e16, E_);

    for (int l = 0; l < 3; ++l) {
        int b0 = 13 + 13 * l;
        k_zero<<<2048, 256, 0, stream>>>(aggr, (size_t)N_ * 128);
        k_edge_msg<<<Etiles, 256, 0, stream>>>(e16, fLE[l], (const float*)d_in[b0 + 2],
                                               src, dst, h, aggr, E_);
        k_node_update<<<Ntiles, 256, 0, stream>>>(h, aggr, (const float*)d_in[b0 + 0],
            fM1[l], (const float*)d_in[b0 + 4], (const float*)d_in[b0 + 5],
            (const float*)d_in[b0 + 6], (const float*)d_in[b0 + 7],
            (const float*)d_in[b0 + 8], fM2[l], (const float*)d_in[b0 + 10],
            (const float*)d_in[b0 + 11], (const float*)d_in[b0 + 12],
            outs + (size_t)l * N_ * 128, N_);
    }

    k_zero<<<64, 256, 0, stream>>>(psum, (size_t)G_ * 128);
    k_zero<<<64, 256, 0, stream>>>(pmax, (size_t)G_ * 128);
    k_zero<<<1, 64, 0, stream>>>(pcnt, (size_t)G_);
    k_jk_pool<<<Ntiles, 256, 0, stream>>>(outs, batch, fJK1,
        (const float*)d_in[53], (const float*)d_in[54], (const float*)d_in[55],
        psum, pmax, pcnt, N_);

    k_final<<<G_, 128, 0, stream>>>(psum, pmax, pcnt, gfeat,
        (const float*)d_in[56], (const float*)d_in[57], (const float*)d_in[58],
        (const float*)d_in[59], (const float*)d_in[60], (const float*)d_in[61],
        (const float*)d_in[62], (const float*)d_in[63], (const float*)d_in[64],
        (const float*)d_in[65], (float*)d_out);
}
